// Prefill_54116587930004
// MI455X (gfx1250) — matrix operand order repaired and checked by reading, compiled, not run on hardware
//
#include <hip/hip_runtime.h>
#include <hip/hip_bf16.h>
#include <math.h>

typedef __attribute__((ext_vector_type(16))) _Float16 v16h;
typedef __attribute__((ext_vector_type(8)))  _Float16 v8h;
typedef __attribute__((ext_vector_type(8)))  float    v8f;

#define BN_INV 0.99999500003749969f   /* 1/sqrt(1+1e-5) */
#define NPTS   12288                  /* 96*128 */
#define NCOLS  49152                  /* NPTS * NUM(4) */

__device__ __forceinline__ float gelu_f(float x) {
  return 0.5f * x * (1.0f + erff(x * 0.7071067811865475f));
}

__device__ __forceinline__ v8f wmma16(v16h a, v16h b, v8f c) {
  return __builtin_amdgcn_wmma_f32_16x16x32_f16(false, a, false, b, (short)0, c,
                                                false, false);
}

// Swizzled tile element index (within one 512-half tile = [lane][16 halves]).
// A-tile (16x32 MxK), CDNA5 ISA 7.12.2: lane = M + 16*((K>>3)&1),
//                                       h = (K&7) + 8*((K>>4)&1).
__device__ __forceinline__ int swzA(int m, int kk) {
  return ((m + (((kk >> 3) & 1) << 4)) << 4) + (kk & 7) + (((kk >> 4) & 1) << 3);
}
// B-tile (32x16 KxN): lane = N + 16*((K>>3)&1), h = (K&7) + 8*((K>>4)&1).
__device__ __forceinline__ int swzB(int n, int kk) {
  return ((n + (((kk >> 3) & 1) << 4)) << 4) + (kk & 7) + (((kk >> 4) & 1) << 3);
}

// One-lane tile fetch: 32 contiguous bytes -> v16h (2x b128 loads).
__device__ __forceinline__ v16h ldtile(const _Float16* __restrict__ p,
                                       int tile, int lane) {
  return *(const v16h*)(p + (((size_t)tile << 5) + lane) * 16);
}

// ---------------- K1: 3x3 conv + BN + ReLU on 12x16 map ----------------
__global__ __launch_bounds__(256) void k_conv3x3(
    const float* __restrict__ fout, const float* __restrict__ Wp,
    const float* __restrict__ gp, const float* __restrict__ bp,
    float* __restrict__ xP) {
  int g = blockIdx.x * 256 + threadIdx.x;          // 128*12*16 = 24576
  int o = g / 192, p = g - o * 192;
  int y = p >> 4, x = p & 15;
  float acc = 0.f;
  for (int ci = 0; ci < 64; ++ci) {
    const float* fin = fout + ci * 192;
    const float* wk = Wp + (o * 64 + ci) * 9;
#pragma unroll
    for (int ky = 0; ky < 3; ++ky) {
      int yy = y + ky - 1;
      if (yy < 0 || yy >= 12) continue;
#pragma unroll
      for (int kx = 0; kx < 3; ++kx) {
        int xx = x + kx - 1;
        if (xx < 0 || xx >= 16) continue;
        acc += fin[yy * 16 + xx] * wk[ky * 3 + kx];
      }
    }
  }
  float v = gp[o] * BN_INV * acc + bp[o];
  xP[g] = v > 0.f ? v : 0.f;
}

// ---------------- K2: bilinear 12x16 -> 96x128, point-major f16 --------
__global__ __launch_bounds__(256) void k_bilinear(
    const float* __restrict__ xP, _Float16* __restrict__ fi_h) {
  int g = blockIdx.x * 256 + threadIdx.x;          // NPTS*128
  int n = g >> 7, c = g & 127;
  int iy = n >> 7, ix = n & 127;
  float ry = (float)iy * (11.0f / 95.0f);
  float rx = (float)ix * (15.0f / 127.0f);
  int y0 = (int)floorf(ry), x0 = (int)floorf(rx);
  int y1 = y0 + 1 < 11 ? y0 + 1 : 11;
  int x1 = x0 + 1 < 15 ? x0 + 1 : 15;
  float wy = ry - (float)y0, wx = rx - (float)x0;
  const float* P = xP + c * 192;
  float a = P[y0 * 16 + x0], b = P[y0 * 16 + x1];
  float cc = P[y1 * 16 + x0], d = P[y1 * 16 + x1];
  float top = a * (1.f - wx) + b * wx;
  float bot = cc * (1.f - wx) + d * wx;
  fi_h[(size_t)n * 128 + c] = (_Float16)(top * (1.f - wy) + bot * wy);
}

// ---------------- K3: brute-force 4-NN with penalty --------------------
__global__ __launch_bounds__(256) void k_knn(
    const float* __restrict__ Sp, int* __restrict__ idx,
    float* __restrict__ nb) {
  __shared__ unsigned vb[384];                     // 12288-bit valid mask
  int tid = threadIdx.x;
  for (int d = tid; d < 384; d += 256) {
    unsigned m = 0;
    int base = d << 5;
    for (int b = 0; b < 32; ++b)
      if (Sp[base + b] > 0.001f) m |= (1u << b);
    vb[d] = m;
  }
  __syncthreads();
  int q = blockIdx.x * 256 + tid;
  int qx = q & 127, qy = q >> 7;
  float bd0 = 3e38f, bd1 = 3e38f, bd2 = 3e38f, bd3 = 3e38f;
  int bi0 = 0, bi1 = 0, bi2 = 0, bi3 = 0;
  for (int j = 0; j < NPTS; ++j) {
    int dx = (j & 127) - qx, dy = (j >> 7) - qy;
    float pen = ((vb[j >> 5] >> (j & 31)) & 1u) ? 0.f : 1e10f;
    float d = (float)(dx * dx + dy * dy) + pen;    // exact ints + pen (f32)
    if (d < bd3) {                                  // strict <: stable ties
      if (d < bd2) {
        bd3 = bd2; bi3 = bi2;
        if (d < bd1) {
          bd2 = bd1; bi2 = bi1;
          if (d < bd0) { bd1 = bd0; bi1 = bi0; bd0 = d; bi0 = j; }
          else         { bd1 = d;   bi1 = j; }
        } else { bd2 = d; bi2 = j; }
      } else { bd3 = d; bi3 = j; }
    }
  }
  int bi[4] = {bi0, bi1, bi2, bi3};
#pragma unroll
  for (int k = 0; k < 4; ++k) {
    int e = q * 4 + k;
    idx[e] = bi[k];
    nb[e * 3 + 0] = Sp[bi[k]];
    nb[e * 3 + 1] = (float)((bi[k] & 127) - qx);
    nb[e * 3 + 2] = (float)((bi[k] >> 7) - qy);
  }
}

// -------- K4: weights -> f16 in WMMA B-register image, K pad 288 -------
// Layout: tiles of 32x16 (KxN): [kt*NT + nt][lane][16 halves].
__global__ __launch_bounds__(256) void k_wprep(
    const float* __restrict__ W1, const float* __restrict__ W2,
    const float* __restrict__ W3, const float* __restrict__ W4,
    const float* __restrict__ Wc,
    _Float16* __restrict__ W1s, _Float16* __restrict__ W2s,
    _Float16* __restrict__ W3s, _Float16* __restrict__ W4s,
    _Float16* __restrict__ Wcs) {
  int g = blockIdx.x * 256 + threadIdx.x;          // up to 288*128 = 36864
  if (g < 288 * 128) {
    int k = g >> 7, o = g & 127;
    int dst = ((k >> 5) * 8 + (o >> 4)) * 512 + swzB(o & 15, k & 31);
    W1s[dst] = (_Float16)(k < 259 ? W1[o * 259 + k] : 0.f);
  }
  if (g < 128 * 128) {
    int k = g >> 7, o = g & 127;
    int dst = ((k >> 5) * 8 + (o >> 4)) * 512 + swzB(o & 15, k & 31);
    W2s[dst] = (_Float16)W2[o * 128 + k];
    W3s[dst] = (_Float16)W3[o * 128 + k];
    W4s[dst] = (_Float16)W4[o * 128 + k];
  }
  if (g < 128 * 16) {
    int k = g >> 4, col = g & 15;
    int dst = (k >> 5) * 512 + swzB(col, k & 31);
    Wcs[dst] = (_Float16)(col < 3 ? Wc[col * 128 + k] : 0.f);
  }
}

// ---------------- K5 (G1): gather X, layer1 + layer2 (WMMA) ------------
// 64 columns/block; LDS tiles stored in WMMA A-register image.
__global__ __launch_bounds__(256) void k_g1(
    const _Float16* __restrict__ fi_h, const int* __restrict__ idx,
    const float* __restrict__ nb, const _Float16* __restrict__ W1s,
    const _Float16* __restrict__ W2s,
    const float* __restrict__ g1, const float* __restrict__ b1,
    const float* __restrict__ g2, const float* __restrict__ b2,
    _Float16* __restrict__ h2s) {
  __shared__ __align__(32) _Float16 As[4 * 9 * 512];   // X^T, 4 Mtiles x 9 Ktiles
  __shared__ __align__(32) _Float16 H1[4 * 4 * 512];   // layer1 out, 4x4 tiles
  const int tid = threadIdx.x;
  const int j0 = blockIdx.x * 64;
  // Build X^T directly in A-register image, 16B at a time: any aligned
  // 8-wide K-octet is contiguous within one lane's 16-half run.
  {
    int j = tid >> 2;
    int J = j0 + j, n = J >> 2, kq = J & 3;
    int gi = idx[n * 4 + kq];
    const _Float16* s0 = fi_h + (size_t)n * 128;
    const _Float16* s1 = fi_h + (size_t)gi * 128;
    const float* nb3 = nb + (size_t)(n * 4 + kq) * 3;
    int mtile = j >> 4, m = j & 15;
    const v8h vzero = (v8h){(_Float16)0.f, (_Float16)0.f, (_Float16)0.f,
                            (_Float16)0.f, (_Float16)0.f, (_Float16)0.f,
                            (_Float16)0.f, (_Float16)0.f};
    for (int it = tid & 3; it < 36; it += 4) {       // 9 Ktiles x 4 octets
      int kt = it >> 2, oct = it & 3;
      int c0 = kt * 32 + oct * 8;
      _Float16* dst = As + (mtile * 9 + kt) * 512 +
                      (m + ((oct & 1) << 4)) * 16 + ((oct >> 1) << 3);
      v8h val;
      if (c0 < 128)       val = *(const v8h*)(s0 + c0);
      else if (c0 < 256)  val = *(const v8h*)(s1 + (c0 - 128));
      else if (c0 == 256) val = (v8h){(_Float16)nb3[0], (_Float16)nb3[1],
                                      (_Float16)nb3[2], (_Float16)0.f,
                                      (_Float16)0.f, (_Float16)0.f,
                                      (_Float16)0.f, (_Float16)0.f};
      else                val = vzero;
      *(v8h*)dst = val;
    }
  }
  __syncthreads();
  const int wave = tid >> 5, lane = tid & 31;
  const int mtile = wave >> 1;                     // 4 M-tiles of 16 points
  const int ntb = (wave & 1) * 4;                  // 4 N-tiles of 16 channels
  const int ncol = lane & 15, mb = (lane >> 4) * 8;
  // ---- layer1: K = 288 (9 tiles) ----
  v8f acc[4];
#pragma unroll
  for (int t = 0; t < 4; ++t) acc[t] = (v8f){0.f,0.f,0.f,0.f,0.f,0.f,0.f,0.f};
  for (int kt = 0; kt < 9; ++kt) {
    v16h a = ldtile(As, mtile * 9 + kt, lane);
#pragma unroll
    for (int t = 0; t < 4; ++t) {
      v16h b = ldtile(W1s, kt * 8 + ntb + t, lane);
      acc[t] = wmma16(a, b, acc[t]);
    }
  }
#pragma unroll
  for (int t = 0; t < 4; ++t) {
    int o = (ntb + t) * 16 + ncol;
    float gg = g1[o] * BN_INV, bb = b1[o];
    int kt2 = o >> 5, kk = o & 31;
#pragma unroll
    for (int v = 0; v < 8; ++v)
      H1[(mtile * 4 + kt2) * 512 + swzA(mb + v, kk)] =
          (_Float16)gelu_f(gg * acc[t][v] + bb);
  }
  __syncthreads();
  // ---- layer2: K = 128 (4 tiles) ----
  v8f acc2[4];
#pragma unroll
  for (int t = 0; t < 4; ++t) acc2[t] = (v8f){0.f,0.f,0.f,0.f,0.f,0.f,0.f,0.f};
  for (int kt = 0; kt < 4; ++kt) {
    v16h a = ldtile(H1, mtile * 4 + kt, lane);
#pragma unroll
    for (int t = 0; t < 4; ++t) {
      v16h b = ldtile(W2s, kt * 8 + ntb + t, lane);
      acc2[t] = wmma16(a, b, acc2[t]);
    }
  }
  // epilogue -> h2 in global, already in A-register image for G2.
  const int gmt = (j0 >> 4) + mtile;
#pragma unroll
  for (int t = 0; t < 4; ++t) {
    int o = (ntb + t) * 16 + ncol;
    float gg = g2[o] * BN_INV, bb = b2[o];
    int kt2 = o >> 5, kk = o & 31;
#pragma unroll
    for (int v = 0; v < 8; ++v)
      h2s[(size_t)(gmt * 4 + kt2) * 512 + swzA(mb + v, kk)] =
          (_Float16)gelu_f(gg * acc2[t][v] + bb);
  }
}

// ------- K6 (G2): layer3+4 + residual, Wc head, softmax reduce ---------
__global__ __launch_bounds__(256) void k_g2(
    const _Float16* __restrict__ h2s, const _Float16* __restrict__ W3s,
    const _Float16* __restrict__ W4s, const _Float16* __restrict__ Wcs,
    const float* __restrict__ g3, const float* __restrict__ b3,
    const float* __restrict__ g4, const float* __restrict__ b4,
    const float* __restrict__ bc, const float* __restrict__ nb,
    float* __restrict__ out) {
  __shared__ __align__(32) _Float16 H3[4 * 4 * 512];
  __shared__ __align__(32) _Float16 H4[4 * 4 * 512];
  __shared__ float FEAT[64 * 4];
  const int tid = threadIdx.x;
  const int j0 = blockIdx.x * 64;
  const int wave = tid >> 5, lane = tid & 31;
  const int mtile = wave >> 1;
  const int ntb = (wave & 1) * 4;
  const int ncol = lane & 15, mb = (lane >> 4) * 8;
  const int gmt = (j0 >> 4) + mtile;
  // ---- layer3: A straight from global h2 (A-register image) ----
  v8f acc[4];
#pragma unroll
  for (int t = 0; t < 4; ++t) acc[t] = (v8f){0.f,0.f,0.f,0.f,0.f,0.f,0.f,0.f};
  for (int kt = 0; kt < 4; ++kt) {
    v16h a = ldtile(h2s, gmt * 4 + kt, lane);
#pragma unroll
    for (int t = 0; t < 4; ++t) {
      v16h b = ldtile(W3s, kt * 8 + ntb + t, lane);
      acc[t] = wmma16(a, b, acc[t]);
    }
  }
#pragma unroll
  for (int t = 0; t < 4; ++t) {
    int o = (ntb + t) * 16 + ncol;
    float gg = g3[o] * BN_INV, bb = b3[o];
    int kt2 = o >> 5, kk = o & 31;
#pragma unroll
    for (int v = 0; v < 8; ++v)
      H3[(mtile * 4 + kt2) * 512 + swzA(mb + v, kk)] =
          (_Float16)gelu_f(gg * acc[t][v] + bb);
  }
  __syncthreads();
  // ---- layer4 + residual + gelu ----
  v8f acc2[4];
#pragma unroll
  for (int t = 0; t < 4; ++t) acc2[t] = (v8f){0.f,0.f,0.f,0.f,0.f,0.f,0.f,0.f};
  for (int kt = 0; kt < 4; ++kt) {
    v16h a = ldtile(H3, mtile * 4 + kt, lane);
#pragma unroll
    for (int t = 0; t < 4; ++t) {
      v16h b = ldtile(W4s, kt * 8 + ntb + t, lane);
      acc2[t] = wmma16(a, b, acc2[t]);
    }
  }
#pragma unroll
  for (int t = 0; t < 4; ++t) {
    int o = (ntb + t) * 16 + ncol;
    float gg = g4[o] * BN_INV, bb = b4[o];
    int kt2 = o >> 5, kk = o & 31;
#pragma unroll
    for (int v = 0; v < 8; ++v) {
      int e = swzA(mb + v, kk);
      float xf = (float)h2s[(size_t)(gmt * 4 + kt2) * 512 + e];
      H4[(mtile * 4 + kt2) * 512 + e] =
          (_Float16)gelu_f(xf + gg * acc2[t][v] + bb);
    }
  }
  __syncthreads();
  // ---- head: Wc (K=128 -> N=16, cols 0..2 valid) ----
  if (wave < 4) {
    v8f ah = (v8f){0.f,0.f,0.f,0.f,0.f,0.f,0.f,0.f};
    for (int kt = 0; kt < 4; ++kt) {
      v16h a = ldtile(H4, wave * 4 + kt, lane);
      v16h b = ldtile(Wcs, kt, lane);
      ah = wmma16(a, b, ah);
    }
    if (ncol < 3) {
      float bcc = bc[ncol];
#pragma unroll
      for (int v = 0; v < 8; ++v)
        FEAT[(wave * 16 + mb + v) * 4 + ncol] = ah[v] + bcc;
    }
  }
  __syncthreads();
  // ---- softmax over the 4 neighbor columns of each point, reduce ------
  if (tid < 16) {
    int n = blockIdx.x * 16 + tid;                 // 16 points per block
    float al[4], be[4], om[4];
#pragma unroll
    for (int k = 0; k < 4; ++k) {
      al[k] = FEAT[(tid * 4 + k) * 4 + 0];
      be[k] = FEAT[(tid * 4 + k) * 4 + 1];
      om[k] = FEAT[(tid * 4 + k) * 4 + 2];
    }
    float mx = om[0];
#pragma unroll
    for (int k = 1; k < 4; ++k) mx = om[k] > mx ? om[k] : mx;
    float e[4], s = 0.f;
#pragma unroll
    for (int k = 0; k < 4; ++k) { e[k] = expf(om[k] - mx); s += e[k]; }
    float dout = 0.f;
#pragma unroll
    for (int k = 0; k < 4; ++k) {
      float pf = nb[(size_t)(n * 4 + k) * 3 + 0];
      dout += ((al[k] + 1.f) * pf + be[k]) * (e[k] / s);
    }
    out[n] = dout;
  }
}

extern "C" void kernel_launch(void* const* d_in, const int* in_sizes, int n_in,
                              void* d_out, int out_size, void* d_ws,
                              size_t ws_size, hipStream_t stream) {
  (void)in_sizes; (void)n_in; (void)out_size; (void)ws_size;
  const float* Sp   = (const float*)d_in[0];
  const float* fout = (const float*)d_in[1];
  const float* Wp   = (const float*)d_in[2];
  const float* gp   = (const float*)d_in[3];
  const float* bp   = (const float*)d_in[4];
  const float* W1   = (const float*)d_in[5];
  const float* g1   = (const float*)d_in[6];
  const float* b1   = (const float*)d_in[7];
  const float* W2   = (const float*)d_in[8];
  const float* g2   = (const float*)d_in[9];
  const float* b2   = (const float*)d_in[10];
  const float* W3   = (const float*)d_in[11];
  const float* g3   = (const float*)d_in[12];
  const float* b3   = (const float*)d_in[13];
  const float* W4   = (const float*)d_in[14];
  const float* g4   = (const float*)d_in[15];
  const float* b4   = (const float*)d_in[16];
  const float* Wc   = (const float*)d_in[17];
  const float* bc   = (const float*)d_in[18];
  float* out = (float*)d_out;

  char* ws = (char*)d_ws;
  size_t off = 0;
  auto carve = [&](size_t bytes) -> void* {
    void* p = ws + off;
    off = (off + bytes + 255) & ~(size_t)255;
    return p;
  };
  float*    xP   = (float*)carve(128 * 192 * sizeof(float));
  _Float16* fi_h = (_Float16*)carve((size_t)NPTS * 128 * sizeof(_Float16));
  int*      idx  = (int*)carve((size_t)NPTS * 4 * sizeof(int));
  float*    nb   = (float*)carve((size_t)NPTS * 4 * 3 * sizeof(float));
  _Float16* W1s  = (_Float16*)carve(288 * 128 * sizeof(_Float16));
  _Float16* W2s  = (_Float16*)carve(128 * 128 * sizeof(_Float16));
  _Float16* W3s  = (_Float16*)carve(128 * 128 * sizeof(_Float16));
  _Float16* W4s  = (_Float16*)carve(128 * 128 * sizeof(_Float16));
  _Float16* Wcs  = (_Float16*)carve(128 * 16 * sizeof(_Float16));
  _Float16* h2s  = (_Float16*)carve((size_t)NCOLS * 128 * sizeof(_Float16));

  k_conv3x3<<<96, 256, 0, stream>>>(fout, Wp, gp, bp, xP);
  k_bilinear<<<6144, 256, 0, stream>>>(xP, fi_h);
  k_knn<<<48, 256, 0, stream>>>(Sp, idx, nb);
  k_wprep<<<144, 256, 0, stream>>>(W1, W2, W3, W4, Wc, W1s, W2s, W3s, W4s, Wcs);
  k_g1<<<NCOLS / 64, 256, 0, stream>>>(fi_h, idx, nb, W1s, W2s, g1, b1, g2, b2,
                                       h2s);
  k_g2<<<NCOLS / 64, 256, 0, stream>>>(h2s, W3s, W4s, Wcs, g3, b3, g4, b4, bc,
                                       nb, out);
}
